// RNN_44315472560514
// MI455X (gfx1250) — matrix operand order repaired and checked by reading, not compiled, not run on hardware
//
#include <hip/hip_runtime.h>

// ---------------------------------------------------------------------------
// Sequential tanh RNN, B=64, T=1024, D_in=D_out=512.
//   * batch rows independent -> 4 WGs, each owns a 16-row slab of h, runs the
//     whole T-step chain with no inter-WG sync.
//   * per step: [x_t | h] (16x1024) @ [[W];[Wr]] (1024x512) as ONE K=1024 GEMM
//     via v_wmma_f32_16x16x32_bf16 (f32 accumulate), + bias, + tanh.
//   * 512 threads/WG = 16 waves @ 4/SIMD -> ~256 VGPR/wave budget; each wave
//     owns TWO 16-column output tiles (n0, n0+256) sharing A fragments.
//   * B (weights, bf16, pre-transposed, 1 MB L2-hot) software-pipelined
//     2 groups deep (4 K-slices/group); issues are UNCONDITIONAL with wrapped
//     (in-bounds) addresses so loads write straight into the loop-carried
//     registers -> no phi copies, no spills. A comes from LDS just-in-time.
//   * group loop keeps a runtime trip count -> compiler cannot hoist the
//     t-invariant weight loads out of the t-loop and spill them.
// ---------------------------------------------------------------------------

typedef __attribute__((ext_vector_type(16))) __bf16 v16bf;
typedef __attribute__((ext_vector_type(8)))  __bf16 v8bf;
typedef __attribute__((ext_vector_type(8)))  float  v8f;
typedef __attribute__((ext_vector_type(4)))  float  v4f;

#define T_STEPS 1024
#define D 512          // D_in == D_out
#define KTOT 1024      // D_in + D_out (fused K)
#define RS 1544        // LDS row stride (bf16): 512 x | 512 h0 | 512 h1 | 8 pad
#define XOFF 0
#define HOFF 512

// --- prep: WT2[n*1024 + k] = bf16( k<512 ? W[k][n] : Wr[k-512][n] ) ----------
__global__ void rnn_prep_wt2(const float* __restrict__ W,
                             const float* __restrict__ Wr,
                             __bf16* __restrict__ WT2) {
  int gid = blockIdx.x * blockDim.x + threadIdx.x;   // 0 .. 512*1024-1
  int n = gid & 511;           // fastest -> coalesced weight reads
  int k = gid >> 9;            // 0..1023
  float v = (k < 512) ? W[(size_t)k * 512 + n] : Wr[(size_t)(k - 512) * 512 + n];
  WT2[(size_t)n * 1024 + k] = (__bf16)v;
}

// one in-flight WEIGHT load group: 4 K-slices x 2 column tiles (B only)
struct Grp {
  v16bf bA[4];
  v16bf bB[4];
};

// --- persistent per-batch-slab RNN kernel: 4 blocks x 512 threads -----------
__global__ __launch_bounds__(512, 1) void rnn_persist(
    const float* __restrict__ x,      // [64][1024][512] f32
    const float* __restrict__ bias,   // [512] f32
    const __bf16* __restrict__ WT2,   // [512][1024] bf16 (transposed W|Wr)
    float* __restrict__ out,          // [64][512] f32
    int nslices) {                    // runtime 32: keeps loops un-hoistable
  __shared__ __bf16 sh[16 * RS];

  const int tid  = threadIdx.x;
  const int wave = tid >> 5;          // 0..15
  const int lane = tid & 31;
  const int lm   = lane & 15;
  const int hi   = lane >> 4;         // 0 | 1 (lane half)
  const int hi8  = hi * 8;
  const int b0   = blockIdx.x * 16;   // batch row base for this WG
  const int n0a  = wave * 16;         // first column tile
  const int n0b  = n0a + 256;         // second column tile

  // A-fragment per-lane LDS bases (row = lm), 16-bit A 16x32 layout
  const __bf16* abase0 = sh + lm * RS + hi * 16;
  const __bf16* abase1 = sh + lm * RS + hi * 16 + 8;

  // zero both h buffers (h_{-1} = 0)
  for (int r = 0; r < 16; ++r) {
    sh[r * RS + HOFF + tid]       = (__bf16)0.0f;
    sh[r * RS + HOFF + 512 + tid] = (__bf16)0.0f;
  }

  const float bvalA = bias[n0a + lm];
  const float bvalB = bias[n0b + lm];
  const __bf16* bbaseA = WT2 + (size_t)(n0a + lm) * KTOT + hi * 16;
  const __bf16* bbaseB = bbaseA + (size_t)256 * KTOT;

  int p = 0;
  for (int t = 0; t < T_STEPS; ++t) {
    // ---- stage x_t slab (16x512 f32 -> bf16), 16 elems / thread ----
#pragma unroll
    for (int half = 0; half < 2; ++half) {
      const int idx  = tid + half * 512;
      const int srow = idx >> 6;
      const int skol = (idx & 63) * 8;
      const v4f* src =
          (const v4f*)(x + ((size_t)(b0 + srow) * T_STEPS + t) * D + skol);
      v4f u = __builtin_nontemporal_load(src);
      v4f v = __builtin_nontemporal_load(src + 1);
      v8bf pk;
      pk[0] = (__bf16)u[0]; pk[1] = (__bf16)u[1]; pk[2] = (__bf16)u[2]; pk[3] = (__bf16)u[3];
      pk[4] = (__bf16)v[0]; pk[5] = (__bf16)v[1]; pk[6] = (__bf16)v[2]; pk[7] = (__bf16)v[3];
      *(v8bf*)&sh[srow * RS + XOFF + skol] = pk;
    }
    __syncthreads();   // x staged + previous h writes visible

    v8f cA, cB;
#pragma unroll
    for (int i = 0; i < 8; ++i) { cA[i] = bvalA; cB[i] = bvalB; }

    const int hextra = HOFF + p * 512 - 512;   // extra LDS offset for h slices

    // issue weight loads for one group of 4 K-slices; offsets wrap so
    // over-prefetch past the last group stays in-bounds (never consumed)
    auto issue = [&](int s, Grp& g) {
#pragma unroll
      for (int j = 0; j < 4; ++j) {
        const int koff = ((s + j) * 32) & (KTOT - 1);
        g.bA[j] = *(const v16bf*)(bbaseA + koff);
        g.bB[j] = *(const v16bf*)(bbaseB + koff);
      }
    };
    // consume one group: A just-in-time from LDS, 8 WMMAs (2 tiles x 4 slices)
    auto consume = [&](int s, Grp& g) {
#pragma unroll
      for (int j = 0; j < 4; ++j) {
        const int ss   = s + j;
        const int loff = ss * 32 + ((ss >= 16) ? hextra : 0);
        v8bf a0 = *(const v8bf*)(abase0 + loff);
        v8bf a1 = *(const v8bf*)(abase1 + loff);
        v16bf a;
#pragma unroll
        for (int k = 0; k < 8; ++k) { a[k] = a0[k]; a[k + 8] = a1[k]; }
        cA = __builtin_amdgcn_wmma_f32_16x16x32_bf16(false, a, false, g.bA[j],
                                                     (short)0, cA, false, false);
        cB = __builtin_amdgcn_wmma_f32_16x16x32_bf16(false, a, false, g.bB[j],
                                                     (short)0, cB, false, false);
      }
    };

    // ---- software-pipelined K=1024 loop: 8 groups, 2 B-groups in flight ----
    const int ngroups = nslices >> 2;          // runtime 8
    Grp g0, g1;
    issue(0, g0);
    issue(4, g1);
    for (int grp = 0; grp < ngroups; grp += 2) {
      consume(grp * 4, g0);
      issue((grp + 2) * 4, g0);                // unconditional, wrapped
      consume((grp + 1) * 4, g1);
      issue((grp + 3) * 4, g1);                // unconditional, wrapped
    }

    // ---- tanh, write h_new into the other buffer; final step -> d_out ----
    const int wb = HOFF + (1 - p) * 512;
#pragma unroll
    for (int r = 0; r < 8; ++r) {
      const int m = r + hi8;                   // C layout: VGPR r -> M = r + 8*hi
#if __has_builtin(__builtin_amdgcn_tanhf)
      float hvA = __builtin_amdgcn_tanhf(cA[r]);   // V_TANH_F32 trans op
      float hvB = __builtin_amdgcn_tanhf(cB[r]);
#else
      float hvA = tanhf(cA[r]);
      float hvB = tanhf(cB[r]);
#endif
      sh[m * RS + wb + n0a + lm] = (__bf16)hvA;
      sh[m * RS + wb + n0b + lm] = (__bf16)hvB;
      if (t == T_STEPS - 1) {
        out[(size_t)(b0 + m) * D + n0a + lm] = hvA;
        out[(size_t)(b0 + m) * D + n0b + lm] = hvB;
      }
    }
    __syncthreads();   // h_new writes done before next step reads/overwrites
    p ^= 1;
  }
}

// ---------------------------------------------------------------------------
extern "C" void kernel_launch(void* const* d_in, const int* in_sizes, int n_in,
                              void* d_out, int out_size, void* d_ws, size_t ws_size,
                              hipStream_t stream) {
  const float* x  = (const float*)d_in[0];   // [64,1024,512]
  const float* W  = (const float*)d_in[1];   // [512,512]
  const float* Wr = (const float*)d_in[2];   // [512,512]
  const float* b  = (const float*)d_in[3];   // [512]
  float* out      = (float*)d_out;           // [64,512]

  __bf16* WT2 = (__bf16*)d_ws;               // 1024*512*2 = 1 MB scratch

  // build fused transposed bf16 weight matrix (stream-ordered before main)
  rnn_prep_wt2<<<(512 * 1024) / 256, 256, 0, stream>>>(W, Wr, WT2);

  // 4 independent batch slabs; 16 waves each, two 16-col output tiles / wave
  rnn_persist<<<4, 512, 0, stream>>>(x, b, WT2, out, 32);
}
